// ChunkedAttention_89558658056830
// MI455X (gfx1250) — matrix operand order repaired and checked by reading, not compiled, not run on hardware
//
#include <hip/hip_runtime.h>
#include <hip/hip_bf16.h>
#include <math.h>

#define HIDDEN   1024
#define HEADS    16
#define HEAD_DIM 64
#define SEQ      8192
#define BATCH    2
#define CHUNK    2048
#define OVERLAP  128
#define STEP     (CHUNK - OVERLAP)   // 1920
#define NCHUNK   5
#define QT       2                   // q 16-row tiles per wave (32 q rows)

typedef __attribute__((ext_vector_type(16))) _Float16 v16h;
typedef __attribute__((ext_vector_type(8)))  float    v8f;

// ---- intra-16-lane xor-butterfly reductions -------------------------------
#if __has_builtin(__builtin_amdgcn_permlane16)
__device__ __forceinline__ float permxor(float x, unsigned s0, unsigned s1) {
    return __uint_as_float(__builtin_amdgcn_permlane16(
        __float_as_uint(x), __float_as_uint(x), s0, s1, false, false));
}
__device__ __forceinline__ float redmax16(float x) {
    x = fmaxf(x, permxor(x, 0x67452301u, 0xEFCDAB89u));   // xor 1
    x = fmaxf(x, permxor(x, 0x45670123u, 0xCDEF89ABu));   // xor 2
    x = fmaxf(x, permxor(x, 0x01234567u, 0x89ABCDEFu));   // xor 4
    x = fmaxf(x, permxor(x, 0xFEDCBA98u, 0x76543210u));   // xor 8
    return x;
}
__device__ __forceinline__ float redsum16(float x) {
    x += permxor(x, 0x67452301u, 0xEFCDAB89u);
    x += permxor(x, 0x45670123u, 0xCDEF89ABu);
    x += permxor(x, 0x01234567u, 0x89ABCDEFu);
    x += permxor(x, 0xFEDCBA98u, 0x76543210u);
    return x;
}
#else
__device__ __forceinline__ float redmax16(float x) {
    #pragma unroll
    for (int m = 1; m < 16; m <<= 1) x = fmaxf(x, __shfl_xor(x, m, 32));
    return x;
}
__device__ __forceinline__ float redsum16(float x) {
    #pragma unroll
    for (int m = 1; m < 16; m <<= 1) x += __shfl_xor(x, m, 32);
    return x;
}
#endif

// Zero the output accumulator.
__global__ void ca_zero_kernel(float* __restrict__ out, size_t n) {
    size_t i = (size_t)blockIdx.x * blockDim.x + threadIdx.x;
    if (i < n) out[i] = 0.0f;
}

// Flash-attention over one (batch, head, chunk, 32-row q tile) per wave.
__global__ __launch_bounds__(32)
void ca_flash_kernel(const float* __restrict__ Q,
                     const float* __restrict__ K,
                     const float* __restrict__ V,
                     float* __restrict__ Out) {
    __shared__ _Float16 pt[QT * 16 * 32];   // P tiles: [qt][qrow][kcol]
    __shared__ _Float16 vt[64 * 33];        // V tile transposed: [dim][key], padded

    const int chunk  = blockIdx.y;
    const int start  = chunk * STEP;
    const int L      = min(CHUNK, SEQ - start);
    const int ntiles = L / (16 * QT);
    if ((int)blockIdx.x >= ntiles) return;          // uniform per-wave exit

    const int b  = blockIdx.z / HEADS;
    const int h  = blockIdx.z % HEADS;
    const int ks = max(0, start - OVERLAP);
    const int ke = min(SEQ, start + L + OVERLAP);
    const int Lk = ke - ks;                         // always a multiple of 32

    const int lane = threadIdx.x;
    const int half = lane >> 4;
    const int mrow = lane & 15;

    const float scale = 0.125f;                     // 1/sqrt(HEAD_DIM)

    // ---- Q tiles (QT x 16 x 64) into A-layout operands, prescaled ----
    v16h Aq[QT][2];
    #pragma unroll
    for (int qt = 0; qt < QT; ++qt) {
        const int qrow = start + ((int)blockIdx.x * QT + qt) * 16 + mrow;
        const float* qp = Q + ((size_t)(b * SEQ + qrow)) * HIDDEN + h * HEAD_DIM;
        #pragma unroll
        for (int s = 0; s < 2; ++s) {
            const int d0 = 32 * s + (half ? 8 : 0);
            #pragma unroll
            for (int j = 0; j < 8; ++j) {
                Aq[qt][s][j]     = (_Float16)(qp[d0 + j]      * scale);
                Aq[qt][s][8 + j] = (_Float16)(qp[d0 + 16 + j] * scale);
            }
        }
    }

    v8f o[QT][4] = {};
    float mstat[QT][8], lstat[QT][8];
    #pragma unroll
    for (int qt = 0; qt < QT; ++qt)
        #pragma unroll
        for (int i = 0; i < 8; ++i) { mstat[qt][i] = -INFINITY; lstat[qt][i] = 0.0f; }

    const int nkb = Lk / 32;
    for (int jb = 0; jb < nkb; ++jb) {
        const int kbase = ks + jb * 32;

        if (jb + 1 < nkb) {
            __builtin_prefetch(K + ((size_t)(b * SEQ + kbase + 32 + mrow)) * HIDDEN
                                 + h * HEAD_DIM, 0, 0);
            __builtin_prefetch(V + ((size_t)(b * SEQ + kbase + 32 + mrow)) * HIDDEN
                                 + h * HEAD_DIM, 0, 0);
        }

        // ---- Load K block once as B-layout f16 (shared across all q tiles) ----
        v16h Bk[2][2];
        #pragma unroll
        for (int t = 0; t < 2; ++t) {
            const float* kp = K + ((size_t)(b * SEQ + kbase + 16 * t + mrow)) * HIDDEN
                                + h * HEAD_DIM;
            #pragma unroll
            for (int s = 0; s < 2; ++s) {
                const int d0 = 32 * s + (half ? 8 : 0);
                #pragma unroll
                for (int j = 0; j < 16; ++j) Bk[t][s][j] = (_Float16)kp[d0 + (j < 8 ? j : 8 + j)];
            }
        }

        // ---- Scores + online softmax per q tile ----
        #pragma unroll
        for (int qt = 0; qt < QT; ++qt) {
            v8f st[2];
            #pragma unroll
            for (int t = 0; t < 2; ++t) {
                v8f c = {};
                c = __builtin_amdgcn_wmma_f32_16x16x32_f16(
                        false, Aq[qt][0], false, Bk[t][0], (short)0, c, false, false);
                c = __builtin_amdgcn_wmma_f32_16x16x32_f16(
                        false, Aq[qt][1], false, Bk[t][1], (short)0, c, false, false);
                st[t] = c;
            }
            _Float16* ptq = pt + qt * (16 * 32);
            #pragma unroll
            for (int i = 0; i < 8; ++i) {
                float mx = redmax16(fmaxf(st[0][i], st[1][i]));
                const float mn = fmaxf(mstat[qt][i], mx);
                const float al = __expf(mstat[qt][i] - mn);
                const float p0 = __expf(st[0][i] - mn);
                const float p1 = __expf(st[1][i] - mn);
                const float rs = redsum16(p0 + p1);
                lstat[qt][i] = al * lstat[qt][i] + rs;
                mstat[qt][i] = mn;
                o[qt][0][i] *= al; o[qt][1][i] *= al;
                o[qt][2][i] *= al; o[qt][3][i] *= al;
                const int row = i + (half ? 8 : 0);
                ptq[row * 32 + mrow]      = (_Float16)p0;
                ptq[row * 32 + 16 + mrow] = (_Float16)p1;
            }
        }

        // ---- Stage V block (32 keys x 64 dims) transposed into LDS as f16 ----
        const float* vp = V + ((size_t)(b * SEQ + kbase + lane)) * HIDDEN + h * HEAD_DIM;
        #pragma unroll 8
        for (int d = 0; d < 64; ++d) vt[d * 33 + lane] = (_Float16)vp[d];
        __syncthreads();

        // ---- Load V B-tiles once, apply to every q tile ----
        v16h Bv[4];
        #pragma unroll
        for (int t = 0; t < 4; ++t) {
            const int dim = 16 * t + mrow;
            const int kb0 = half ? 8 : 0;
            #pragma unroll
            for (int j = 0; j < 16; ++j) Bv[t][j] = vt[dim * 33 + kb0 + (j < 8 ? j : 8 + j)];
        }
        #pragma unroll
        for (int qt = 0; qt < QT; ++qt) {
            const _Float16* ptq = pt + qt * (16 * 32);
            v16h Ap;
            const int kb0 = half ? 8 : 0;
            #pragma unroll
            for (int j = 0; j < 8; ++j) {
                Ap[j]     = ptq[mrow * 32 + kb0 + j];
                Ap[8 + j] = ptq[mrow * 32 + kb0 + 16 + j];
            }
            #pragma unroll
            for (int t = 0; t < 4; ++t) {
                o[qt][t] = __builtin_amdgcn_wmma_f32_16x16x32_f16(
                               false, Ap, false, Bv[t], (short)0, o[qt][t], false, false);
            }
        }
        __syncthreads();   // protect LDS tiles before next iteration rewrites them
    }

    // ---- Normalize, apply chunk-merge ramp weight, accumulate into Out ----
    #pragma unroll
    for (int qt = 0; qt < QT; ++qt) {
        #pragma unroll
        for (int i = 0; i < 8; ++i) {
            const int row = i + (half ? 8 : 0);
            const int p   = ((int)blockIdx.x * QT + qt) * 16 + row;
            float w = 1.0f;
            if (chunk > 0 && p < OVERLAP)               w = (float)p / (OVERLAP - 1);
            if (chunk < NCHUNK - 1 && p >= L - OVERLAP) w = (float)(L - 1 - p) / (OVERLAP - 1);
            const float inv = w / lstat[qt][i];
            const int pos = start + p;
            #pragma unroll
            for (int t = 0; t < 4; ++t) {
                const int dim = 16 * t + mrow;
                const float val = o[qt][t][i] * inv;
                unsafeAtomicAdd(
                    &Out[((size_t)(b * SEQ + pos)) * HIDDEN + h * HEAD_DIM + dim], val);
            }
        }
    }
}

// Divide by analytically computed weight sum (each pos covered by <=2 chunks).
__global__ void ca_finalize_kernel(float* __restrict__ out, size_t n) {
    size_t idx = (size_t)blockIdx.x * blockDim.x + threadIdx.x;
    if (idx >= n) return;
    const int pos = (int)((idx / HIDDEN) % SEQ);
    float ws = 0.0f;
    #pragma unroll
    for (int i = 0; i < NCHUNK; ++i) {
        const int start = i * STEP;
        const int L = min(CHUNK, SEQ - start);
        const int p = pos - start;
        if (p >= 0 && p < L) {
            float w = 1.0f;
            if (i > 0 && p < OVERLAP)               w = (float)p / (OVERLAP - 1);
            if (i < NCHUNK - 1 && p >= L - OVERLAP) w = (float)(L - 1 - p) / (OVERLAP - 1);
            ws += w;
        }
    }
    out[idx] = out[idx] / (ws + 1e-10f);
}

extern "C" void kernel_launch(void* const* d_in, const int* in_sizes, int n_in,
                              void* d_out, int out_size, void* d_ws, size_t ws_size,
                              hipStream_t stream) {
    const float* Q = (const float*)d_in[0];
    const float* K = (const float*)d_in[1];
    const float* V = (const float*)d_in[2];
    float* Out = (float*)d_out;

    const size_t n = (size_t)BATCH * SEQ * HIDDEN;
    ca_zero_kernel<<<(unsigned)((n + 255) / 256), 256, 0, stream>>>(Out, n);

    dim3 grid(CHUNK / (16 * QT), NCHUNK, BATCH * HEADS);
    ca_flash_kernel<<<grid, 32, 0, stream>>>(Q, K, V, Out);

    ca_finalize_kernel<<<(unsigned)((n + 255) / 256), 256, 0, stream>>>(Out, n);
}
